// HGTNet_79113297592416
// MI455X (gfx1250) — matrix operand order repaired and checked by reading, not compiled, not run on hardware
//
#include <hip/hip_runtime.h>
#include <hip/hip_bf16.h>
#include <math.h>

// ---------------------------------------------------------------------------
// HANConv (2 layers, 3 edge types) + heads for MI455X / gfx1250.
// Dense GEMMs -> v_wmma_f32_16x16x32_f16 (f32 accumulate), W staged in LDS
// as zero-padded transposed f16 so every fragment load is unconditional.
// Edge softmax/scatter -> global float atomics (L2-resident accumulators).
// ---------------------------------------------------------------------------

#define HID   64
#define HEADS 2
#define DH    32

typedef __attribute__((ext_vector_type(16))) _Float16 v16h;
typedef __attribute__((ext_vector_type(8)))  float    v8f;

union HFrag { v16h v; _Float16 h[16]; };

// ----------------------------- device helpers ------------------------------

__device__ __forceinline__ void atomicMaxF(float* addr, float val) {
  // monotonic int-bits trick; works with -inf init
  if (val >= 0.f) atomicMax((int*)addr, __float_as_int(val));
  else            atomicMin((unsigned int*)addr, (unsigned int)__float_as_int(val));
}

__device__ __forceinline__ void cvt8(HFrag& f, int base, float4 a, float4 b) {
  f.h[base + 0] = (_Float16)a.x; f.h[base + 1] = (_Float16)a.y;
  f.h[base + 2] = (_Float16)a.z; f.h[base + 3] = (_Float16)a.w;
  f.h[base + 4] = (_Float16)b.x; f.h[base + 5] = (_Float16)b.y;
  f.h[base + 6] = (_Float16)b.z; f.h[base + 7] = (_Float16)b.w;
}

// A fragment (16x32 f16, ISA 7.12.2): lane L holds row M=L&15;
// half j<8 -> K = k0 + 8*(L>>4) + j ; half j>=8 -> K = k0+16 + 8*(L>>4) + (j-8)
// K and k0 are compile-time -> conditions fold, loads are b128.
template <int K>
__device__ __forceinline__ v16h load_a_frag(const float* __restrict__ arow,
                                            int k0, int g) {
  HFrag f;
  const float* p = arow + k0 + 8 * g;
  cvt8(f, 0, *(const float4*)p, *(const float4*)(p + 4));
  if (k0 + 16 < K) {
    const float* p2 = arow + k0 + 16 + 8 * g;
    cvt8(f, 8, *(const float4*)p2, *(const float4*)(p2 + 4));
  } else {
#pragma unroll
    for (int j = 0; j < 8; ++j) f.h[8 + j] = (_Float16)0.0f;
  }
  return f.v;
}

// ------------------------------ WMMA GEMMs ---------------------------------
// D[N x 64] = act(A[N x K] @ W[K x 64] + b).
// block=(32,4): wave y owns a 16-col tile; block covers 4 row tiles (64 rows).
// W is staged once per block into LDS, transposed + zero-padded to KP, f16;
// B fragment = two unconditional 16B ds loads per K-chunk, hoisted out of the
// row loop.  MODE: 0 none, 1 relu, 2 leaky(0.01)

template <int MODE, int K>
__global__ void gemm_n64(const float* __restrict__ A, const float* __restrict__ W,
                         const float* __restrict__ bias, float* __restrict__ D,
                         int N) {
  constexpr int KP = ((K + 31) / 32) * 32;   // padded K (multiple of 32)
  constexpr int NCH = KP / 32;               // K-chunks
  __shared__ alignas(32) _Float16 Wt[64][KP];

  int lane = threadIdx.x;
  int tid  = threadIdx.y * 32 + lane;

  // cooperative stage: global-coalesced read, transposed LDS write, zero pad
  for (int idx = tid; idx < KP * 64; idx += 128) {
    int k = idx >> 6, c = idx & 63;
    float v = (k < K) ? W[idx] : 0.f;        // W row-major [K x 64]
    Wt[c][k] = (_Float16)v;
  }
  __syncthreads();

  int g = lane >> 4, ml = lane & 15;
  int col = threadIdx.y * 16 + ml;
  float bv = bias[col];

  // B fragments: lane holds column `col`, K rows [k0+8g, +8) and [k0+16+8g, +8)
  v16h bf[NCH];
#pragma unroll
  for (int cch = 0; cch < NCH; ++cch) {
    HFrag t;
    *(uint4*)&t.h[0] = *(const uint4*)&Wt[col][cch * 32 + 8 * g];
    *(uint4*)&t.h[8] = *(const uint4*)&Wt[col][cch * 32 + 16 + 8 * g];
    bf[cch] = t.v;
  }

  for (int t = 0; t < 4; ++t) {
    int row0 = (blockIdx.x * 4 + t) * 16;
    if (row0 >= N) break;
    int rowA = row0 + ml; if (rowA >= N) rowA = N - 1;
    const float* arow = A + (size_t)rowA * K;

    // prefetch next tile's A rows
    int rown = row0 + 16 + ml;
    if (t < 3 && rown < N) __builtin_prefetch(A + (size_t)rown * K, 0, 1);

    v8f acc = {};
#pragma unroll
    for (int cch = 0; cch < NCH; ++cch) {
      v16h af = load_a_frag<K>(arow, cch * 32, g);
      acc = __builtin_amdgcn_wmma_f32_16x16x32_f16(false, af, false, bf[cch],
                                                   (short)0, acc, false, false);
    }
#pragma unroll
    for (int r = 0; r < 8; ++r) {
      int rowg = row0 + r + 8 * g;           // C/D layout: M = r + 8*(lane>>4)
      if (rowg < N) {
        float v = acc[r] + bv;
        if (MODE == 1) v = fmaxf(v, 0.f);
        if (MODE == 2) v = (v > 0.f) ? v : 0.01f * v;
        D[(size_t)rowg * HID + col] = v;
      }
    }
  }
}

// score += sum_n sum_c tanh((O @ KW + kb)[n,c]) * q[c]   (semantic attention)
// Same staging/tiling as gemm_n64; K = 64 fixed.
__global__ void score_gemm(const float* __restrict__ O, const float* __restrict__ KW,
                           const float* __restrict__ KB, const float* __restrict__ Q,
                           float* __restrict__ score, int N) {
  __shared__ alignas(32) _Float16 Wt[64][64];
  int lane = threadIdx.x;
  int tid  = threadIdx.y * 32 + lane;
  for (int idx = tid; idx < 64 * 64; idx += 128) {
    int k = idx >> 6, c = idx & 63;
    Wt[c][k] = (_Float16)KW[idx];
  }
  __syncthreads();

  int g = lane >> 4, ml = lane & 15;
  int col = threadIdx.y * 16 + ml;
  float kb = KB[col], q = Q[col];

  HFrag t0, t1;
  *(uint4*)&t0.h[0] = *(const uint4*)&Wt[col][8 * g];
  *(uint4*)&t0.h[8] = *(const uint4*)&Wt[col][16 + 8 * g];
  *(uint4*)&t1.h[0] = *(const uint4*)&Wt[col][32 + 8 * g];
  *(uint4*)&t1.h[8] = *(const uint4*)&Wt[col][48 + 8 * g];
  v16h bf0 = t0.v;
  v16h bf1 = t1.v;

  float partial = 0.f;
  for (int t = 0; t < 4; ++t) {
    int row0 = (blockIdx.x * 4 + t) * 16;
    if (row0 >= N) break;
    int rowA = row0 + ml; if (rowA >= N) rowA = N - 1;
    const float* arow = O + (size_t)rowA * HID;

    v8f acc = {};
    v16h a0 = load_a_frag<HID>(arow, 0, g);
    acc = __builtin_amdgcn_wmma_f32_16x16x32_f16(false, a0, false, bf0,
                                                 (short)0, acc, false, false);
    v16h a1 = load_a_frag<HID>(arow, 32, g);
    acc = __builtin_amdgcn_wmma_f32_16x16x32_f16(false, a1, false, bf1,
                                                 (short)0, acc, false, false);
#pragma unroll
    for (int r = 0; r < 8; ++r) {
      int rowg = row0 + r + 8 * g;
      float tv = tanhf(acc[r] + kb) * q;
      partial += (rowg < N) ? tv : 0.f;
    }
  }
#pragma unroll
  for (int off = 16; off > 0; off >>= 1) partial += __shfl_xor(partial, off, 32);
  if (lane == 0) atomicAdd(score, partial);
}

// ----------------------------- small kernels -------------------------------

__global__ void fill_f32(float* __restrict__ p, float v, int n) {
  int i = blockIdx.x * blockDim.x + threadIdx.x;
  if (i < n) p[i] = v;
}

__global__ void relu_buf(float* __restrict__ p, int n) {
  int i = blockIdx.x * blockDim.x + threadIdx.x;
  if (i < n) p[i] = fmaxf(p[i], 0.f);
}

// alpha[n*2+h] = sum_d X[n, h*32+d] * A[h*32+d]
__global__ void node_alpha(const float* __restrict__ X, const float* __restrict__ A,
                           float* __restrict__ out, int N) {
  int t = blockIdx.x * blockDim.x + threadIdx.x;
  if (t >= N * HEADS) return;
  int n = t >> 1, h = t & 1;
  const float* xp = X + (size_t)n * HID + h * DH;
  const float* ap = A + h * DH;
  float s = 0.f;
#pragma unroll
  for (int d = 0; d < DH; ++d) s += xp[d] * ap[d];
  out[t] = s;
}

// pass A: logits = leaky_relu(aS[src]+aD[dst], 0.2); segment max into mbuf
__global__ void edge_logits(const float* __restrict__ aS, const float* __restrict__ aD,
                            const int* __restrict__ src, const int* __restrict__ dst,
                            float* __restrict__ elog, float* __restrict__ mbuf, int E) {
  int e = blockIdx.x * blockDim.x + threadIdx.x;
  if (e >= E) return;
  int s = src[e], d = dst[e];
#pragma unroll
  for (int h = 0; h < HEADS; ++h) {
    float lg = aS[s * 2 + h] + aD[d * 2 + h];
    lg = (lg > 0.f) ? lg : 0.2f * lg;
    elog[(size_t)e * 2 + h] = lg;
    atomicMaxF(&mbuf[d * 2 + h], lg);
  }
}

// pass B: e = exp(logit - m[dst]); segment sum into sbuf
__global__ void edge_exp(float* __restrict__ elog, const float* __restrict__ mbuf,
                         float* __restrict__ sbuf, const int* __restrict__ dst, int E) {
  int e = blockIdx.x * blockDim.x + threadIdx.x;
  if (e >= E) return;
  int d = dst[e];
#pragma unroll
  for (int h = 0; h < HEADS; ++h) {
    float v = __expf(elog[(size_t)e * 2 + h] - mbuf[d * 2 + h]);
    elog[(size_t)e * 2 + h] = v;
    atomicAdd(&sbuf[d * 2 + h], v);
  }
}

// pass C: out[dst] += x_src[src] * attn.  32 lanes/edge, 2 channels per lane.
__global__ void edge_scatter(const float* __restrict__ elog, const float* __restrict__ sbuf,
                             const int* __restrict__ src, const int* __restrict__ dst,
                             const float* __restrict__ hsrc, float* __restrict__ out, int E) {
  long long tid = (long long)blockIdx.x * blockDim.x + threadIdx.x;
  int e = (int)(tid >> 5);
  if (e >= E) return;
  int c = (int)(tid & 31) * 2;      // channel pair 0..62
  int h = c >> 5;                   // head = c/32
  int s = src[e], d = dst[e];
  float attn = elog[(size_t)e * 2 + h] / (sbuf[d * 2 + h] + 1e-16f);
  const float2 v = *(const float2*)(hsrc + (size_t)s * HID + c);
  float* op = out + (size_t)d * HID + c;
  atomicAdd(op + 0, v.x * attn);
  atomicAdd(op + 1, v.y * attn);
}

// x = a0*o0 + a1*o1 with (a0,a1) = softmax(score/N)
__global__ void combine_cell(const float* __restrict__ o0, const float* __restrict__ o1,
                             const float* __restrict__ score, float* __restrict__ out,
                             int n, float invN) {
  int i = blockIdx.x * blockDim.x + threadIdx.x;
  if (i >= n) return;
  float s0 = score[0] * invN, s1 = score[1] * invN;
  float mx = fmaxf(s0, s1);
  float e0 = __expf(s0 - mx), e1 = __expf(s1 - mx);
  float inv = 1.f / (e0 + e1);
  out[i] = (e0 * inv) * o0[i] + (e1 * inv) * o1[i];
}

// accum += sum_n dot(x[n,:], w)   (value head numerator)
__global__ void value_dot(const float* __restrict__ x, const float* __restrict__ w,
                          float* __restrict__ accum, int N) {
  int n = blockIdx.x * blockDim.x + threadIdx.x;
  float d = 0.f;
  if (n < N) {
    const float* xp = x + (size_t)n * HID;
#pragma unroll
    for (int c = 0; c < HID; ++c) d += xp[c] * w[c];
  }
#pragma unroll
  for (int off = 16; off > 0; off >>= 1) d += __shfl_xor(d, off, 32);
  if ((threadIdx.x & 31) == 0) atomicAdd(accum, d);
}

__global__ void finalize_value(const float* __restrict__ accum,
                               const float* __restrict__ bias,
                               float* __restrict__ out, float invN) {
  out[0] = accum[0] * invN + bias[0];
}

// out[n,m] = tanh(b[m] + sum_c h[n,c]*W[c,m]) ; W row-major [64 x M]
__global__ void head_small(const float* __restrict__ hbuf, const float* __restrict__ W,
                           const float* __restrict__ b, float* __restrict__ out,
                           int N, int M) {
  int n = blockIdx.x * blockDim.x + threadIdx.x;
  if (n >= N) return;
  const float* hp = hbuf + (size_t)n * HID;
  for (int m = 0; m < M; ++m) {
    float s = b[m];
#pragma unroll
    for (int c = 0; c < HID; ++c) s += hp[c] * W[c * M + m];
    out[(size_t)n * M + m] = tanhf(s);
  }
}

// ------------------------------- host side ---------------------------------

static inline int cdiv(long long a, long long b) { return (int)((a + b - 1) / b); }

static void launch_gemm(int mode, const float* A, const float* W, const float* b,
                        float* D, int N, int K, hipStream_t st) {
  dim3 grid(cdiv(N, 64)), block(32, 4);
  if (mode == 2) {
    gemm_n64<2, 64><<<grid, block, 0, st>>>(A, W, b, D, N);
  } else if (K == 16) {
    gemm_n64<0, 16><<<grid, block, 0, st>>>(A, W, b, D, N);
  } else if (K == 32) {
    gemm_n64<0, 32><<<grid, block, 0, st>>>(A, W, b, D, N);
  } else {
    gemm_n64<0, 64><<<grid, block, 0, st>>>(A, W, b, D, N);
  }
}

static void run_edge(const int* src, const int* dst, const float* aS, const float* aD,
                     const float* hsrc, float* outbuf, int E, int Ndst,
                     float* elog, float* mbuf, float* sbuf, hipStream_t st) {
  fill_f32<<<cdiv(Ndst * 2, 256), 256, 0, st>>>(mbuf, -INFINITY, Ndst * 2);
  fill_f32<<<cdiv(Ndst * 2, 256), 256, 0, st>>>(sbuf, 0.f, Ndst * 2);
  fill_f32<<<cdiv((long long)Ndst * HID, 256), 256, 0, st>>>(outbuf, 0.f, Ndst * HID);
  edge_logits<<<cdiv(E, 256), 256, 0, st>>>(aS, aD, src, dst, elog, mbuf, E);
  edge_exp<<<cdiv(E, 256), 256, 0, st>>>(elog, mbuf, sbuf, dst, E);
  edge_scatter<<<cdiv((long long)E * 32, 256), 256, 0, st>>>(elog, sbuf, src, dst,
                                                             hsrc, outbuf, E);
  relu_buf<<<cdiv((long long)Ndst * HID, 256), 256, 0, st>>>(outbuf, Ndst * HID);
}

extern "C" void kernel_launch(void* const* d_in, const int* in_sizes, int n_in,
                              void* d_out, int out_size, void* d_ws, size_t ws_size,
                              hipStream_t stream) {
  (void)n_in; (void)out_size; (void)ws_size;

  const float* x_center = (const float*)d_in[0];
  const float* x_cell   = (const float*)d_in[1];
  const int*   e_cc     = (const int*)d_in[2];   // [2, E_cc]  cell->cell
  const int*   e_rc     = (const int*)d_in[3];   // [2, E_rc]  center->cell
  const int*   e_cr     = (const int*)d_in[4];   // [2, E_cr]  cell->center
  const int Ecc = in_sizes[2] / 2, Erc = in_sizes[3] / 2, Ecr = in_sizes[4] / 2;
  const int Nc  = in_sizes[0] / 16;              // 10000 centers
  const int Nl  = in_sizes[1] / 32;              // 100000 cells

  // params dict flattened as jax pytree leaves (sorted key order), starting d_in[5]
  const float* adst_cc[2], *adst_ccr[2], *adst_ctc[2];
  const float* asrc_cc[2], *asrc_ccr[2], *asrc_ctc[2];
  const float* klin_b[2], *klin_w[2], *proj_cell_b[2], *proj_cell_w[2];
  const float* proj_ctr_b[2], *proj_ctr_w[2], *qv[2];
  for (int l = 0; l < 2; ++l) {
    int b = 5 + 13 * l;
    adst_cc[l]     = (const float*)d_in[b + 0];  // l{l}_adst_cell_to_cell
    adst_ccr[l]    = (const float*)d_in[b + 1];  // l{l}_adst_cell_to_center
    adst_ctc[l]    = (const float*)d_in[b + 2];  // l{l}_adst_center_to_cell
    asrc_cc[l]     = (const float*)d_in[b + 3];
    asrc_ccr[l]    = (const float*)d_in[b + 4];
    asrc_ctc[l]    = (const float*)d_in[b + 5];
    klin_b[l]      = (const float*)d_in[b + 6];
    klin_w[l]      = (const float*)d_in[b + 7];
    proj_cell_b[l] = (const float*)d_in[b + 8];
    proj_cell_w[l] = (const float*)d_in[b + 9];
    proj_ctr_b[l]  = (const float*)d_in[b + 10];
    proj_ctr_w[l]  = (const float*)d_in[b + 11];
    qv[l]          = (const float*)d_in[b + 12];
  }
  const float* lin0_b      = (const float*)d_in[31];
  const float* lin0_w      = (const float*)d_in[32];
  const float* linCell_b   = (const float*)d_in[33];
  const float* linCell_w   = (const float*)d_in[34];
  const float* linCenter_b = (const float*)d_in[35];
  const float* linCenter_w = (const float*)d_in[36];
  const float* linV_b      = (const float*)d_in[37];
  const float* linV_w      = (const float*)d_in[38];

  // workspace carve-up (~123 MB, all offsets 16B-aligned: counts are mult of 4)
  float* WS = (float*)d_ws;
  size_t o = 0;
  auto alloc = [&](size_t n) { float* p = WS + o; o += n; return p; };
  float* hc     = alloc((size_t)Nc * HID);   // projected centers [Nc,2,32]
  float* hl     = alloc((size_t)Nl * HID);   // projected cells
  float* outC   = alloc((size_t)Nc * HID);   // center aggregate (== next x_center)
  float* outT   = alloc((size_t)Nl * HID);   // cell aggregate from center_to_cell
  float* outB   = alloc((size_t)Nl * HID);   // cell aggregate from cell_to_cell
  float* xl     = alloc((size_t)Nl * HID);   // combined cell features
  float* aS_ctc = alloc((size_t)Nc * 2);
  float* aD_ctc = alloc((size_t)Nl * 2);
  float* aS_ccr = alloc((size_t)Nl * 2);
  float* aD_ccr = alloc((size_t)Nc * 2);
  float* aS_cc  = alloc((size_t)Nl * 2);
  float* aD_cc  = alloc((size_t)Nl * 2);
  int Emax = Ecc > Erc ? Ecc : Erc; if (Ecr > Emax) Emax = Ecr;
  float* elog = alloc((size_t)Emax * 2);
  float* mbuf = alloc((size_t)Nl * 2);
  float* sbuf = alloc((size_t)Nl * 2);
  float* scal = alloc(4);                    // [score0, score1, value_accum]

  for (int l = 0; l < 2; ++l) {
    const float* xc_in = (l == 0) ? x_center : outC;
    const float* xl_in = (l == 0) ? x_cell   : xl;
    const int Kc = (l == 0) ? 16 : HID;
    const int Kl = (l == 0) ? 32 : HID;

    // node projections (WMMA)
    launch_gemm(0, xc_in, proj_ctr_w[l],  proj_ctr_b[l],  hc, Nc, Kc, stream);
    launch_gemm(0, xl_in, proj_cell_w[l], proj_cell_b[l], hl, Nl, Kl, stream);

    // per-node attention logit halves
    node_alpha<<<cdiv(Nc * 2, 256), 256, 0, stream>>>(hc, asrc_ctc[l], aS_ctc, Nc);
    node_alpha<<<cdiv(Nl * 2, 256), 256, 0, stream>>>(hl, adst_ctc[l], aD_ctc, Nl);
    node_alpha<<<cdiv(Nl * 2, 256), 256, 0, stream>>>(hl, asrc_ccr[l], aS_ccr, Nl);
    node_alpha<<<cdiv(Nc * 2, 256), 256, 0, stream>>>(hc, adst_ccr[l], aD_ccr, Nc);
    node_alpha<<<cdiv(Nl * 2, 256), 256, 0, stream>>>(hl, asrc_cc[l],  aS_cc,  Nl);
    node_alpha<<<cdiv(Nl * 2, 256), 256, 0, stream>>>(hl, adst_cc[l],  aD_cc,  Nl);

    // edge convolutions (segment softmax + scatter), reusing elog/mbuf/sbuf
    run_edge(e_rc, e_rc + Erc, aS_ctc, aD_ctc, hc, outT, Erc, Nl, elog, mbuf, sbuf, stream);
    run_edge(e_cr, e_cr + Ecr, aS_ccr, aD_ccr, hl, outC, Ecr, Nc, elog, mbuf, sbuf, stream);
    run_edge(e_cc, e_cc + Ecc, aS_cc,  aD_cc,  hl, outB, Ecc, Nl, elog, mbuf, sbuf, stream);

    // semantic attention: center has 1 relation -> identity (x_center := outC).
    fill_f32<<<1, 32, 0, stream>>>(scal, 0.f, 2);
    score_gemm<<<cdiv(Nl, 64), dim3(32, 4), 0, stream>>>(outT, klin_w[l], klin_b[l],
                                                         qv[l], scal + 0, Nl);
    score_gemm<<<cdiv(Nl, 64), dim3(32, 4), 0, stream>>>(outB, klin_w[l], klin_b[l],
                                                         qv[l], scal + 1, Nl);
    combine_cell<<<cdiv((long long)Nl * HID, 256), 256, 0, stream>>>(
        outT, outB, scal, xl, Nl * HID, 1.f / (float)Nl);
  }

  float* out = (float*)d_out;

  // value head: mean over concat(x_center, x_cell) of x@w + b
  fill_f32<<<1, 32, 0, stream>>>(scal + 2, 0.f, 1);
  value_dot<<<cdiv(Nc, 256), 256, 0, stream>>>(outC, linV_w, scal + 2, Nc);
  value_dot<<<cdiv(Nl, 256), 256, 0, stream>>>(xl,   linV_w, scal + 2, Nl);

  // shared hidden head, leaky_relu 0.01 (WMMA); reuse hc/hl as h buffers
  launch_gemm(2, outC, lin0_w, lin0_b, hc, Nc, HID, stream);
  launch_gemm(2, xl,   lin0_w, lin0_b, hl, Nl, HID, stream);

  head_small<<<cdiv(Nc, 256), 256, 0, stream>>>(hc, linCenter_w, linCenter_b,
                                                out, Nc, 3);
  head_small<<<cdiv(Nl, 256), 256, 0, stream>>>(hl, linCell_w, linCell_b,
                                                out + (size_t)Nc * 3, Nl, 2);
  finalize_value<<<1, 1, 0, stream>>>(scal + 2, linV_b,
                                      out + (size_t)Nc * 3 + (size_t)Nl * 2,
                                      1.f / (float)(Nc + Nl));
}
